// Model_GRU_68118181314734
// MI455X (gfx1250) — matrix operand order repaired and checked by reading, not compiled, not run on hardware
//
#include <hip/hip_runtime.h>

typedef _Float16 v16h __attribute__((ext_vector_type(16)));
typedef _Float16 v8h  __attribute__((ext_vector_type(8)));
typedef float    v8f  __attribute__((ext_vector_type(8)));
typedef unsigned int u32x4 __attribute__((ext_vector_type(4)));
typedef int          i32x4 __attribute__((ext_vector_type(4)));
typedef int          i32x8 __attribute__((ext_vector_type(8)));

#define B_   512
#define SS_  4
#define T_   64
#define D_   300
#define H_   256
#define G_   768    // 3*H
#define NSEQ 2048   // B*SS
#define KPAD 320    // D padded to multiple of 32
#define KCH  (KPAD / 32)

#if defined(__has_builtin)
# if __has_builtin(__builtin_amdgcn_tensor_load_to_lds)
#  define HAVE_TDM 1
# endif
#endif
#ifndef HAVE_TDM
# define HAVE_TDM 0
#endif

__device__ __forceinline__ float sig_f(float x)  { return 1.0f / (1.0f + __expf(-x)); }
__device__ __forceinline__ float tanh_f(float x) { return 1.0f - 2.0f / (1.0f + __expf(2.0f * x)); }

__device__ __forceinline__ v8f v8f_zero() {
    v8f z;
#pragma unroll
    for (int i = 0; i < 8; ++i) z[i] = 0.0f;
    return z;
}

// 16x32 f16 A-fragment: lane<16 holds row, K {0..15};
// lane>=16 holds row, K {16..31}. rowptr already includes the +16 sub-ofs.
__device__ __forceinline__ v16h make_a_frag(const _Float16* rowptr) {
    v8h lo  = *(const v8h*)(rowptr);
    v8h hi8 = *(const v8h*)(rowptr + 8);
    return __builtin_shufflevector(lo, hi8, 0,1,2,3,4,5,6,7,8,9,10,11,12,13,14,15);
}

#if HAVE_TDM
// Generic 2D TDM tile load of f16 data (data_size=2B). tensor dims are set
// equal to tile dims (no OOB clipping). pad_bits = pad_enable/interval/amount
// field bits for group1 dword0 (or 0 for a packed copy).
__device__ __forceinline__ void tdm_load_2d(void* lds_dst, const void* gsrc,
                                            unsigned tile_d0, unsigned tile_d1,
                                            unsigned stride0, unsigned pad_bits) {
    unsigned long long ga = (unsigned long long)(size_t)gsrc;
    unsigned int lds_off  = (unsigned int)(size_t)lds_dst;   // flat LDS addr[31:0]
    unsigned td1 = tile_d1 ? tile_d1 : 1u;
    u32x4 g0;
    g0[0] = 1u;                                              // count=1, is_restore=0
    g0[1] = lds_off;                                         // lds_addr
    g0[2] = (unsigned int)(ga & 0xffffffffu);                // global_addr[31:0]
    g0[3] = (unsigned int)((ga >> 32) & 0x01ffffffu)         // global_addr[56:32]
          | (2u << 30);                                      // type=2 (image)
    i32x8 g1;
    g1[0] = (int)((1u << 16) | pad_bits);                    // data_size=2B
    g1[1] = (int)((tile_d0 & 0xffffu) << 16);                // tensor_dim0[15:0]
    g1[2] = (int)((tile_d0 >> 16) | ((td1 & 0xffffu) << 16));// tensor_dim0 hi, dim1 lo
    g1[3] = (int)((td1 >> 16) | ((tile_d0 & 0xffffu) << 16));// dim1 hi, tile_dim0
    g1[4] = (int)(tile_d1 & 0xffffu);                        // tile_dim1, tile_dim2=0
    g1[5] = (int)stride0;                                    // tensor_dim0_stride lo
    g1[6] = 0;
    g1[7] = 0;
    i32x4 z4 = {0, 0, 0, 0};
# if __clang_major__ >= 23
    i32x8 z8 = {0, 0, 0, 0, 0, 0, 0, 0};
    __builtin_amdgcn_tensor_load_to_lds(g0, g1, z4, z4, z8, 0);
# else
    __builtin_amdgcn_tensor_load_to_lds(g0, g1, z4, z4, 0);
# endif
}
#define PADBITS_64B_PLUS_16B ((1u << 20) | (3u << 22) | (3u << 25))
#endif

// ---------------------------------------------------------------------------
// Kernel 0a: convert layer-1 weights to f16 (w_ih1 padded K 300->320 with 0)
// ---------------------------------------------------------------------------
__global__ void k_cvt_weights(const float* __restrict__ wih,
                              const float* __restrict__ whh,
                              _Float16* __restrict__ wih16,
                              _Float16* __restrict__ whh16) {
    int idx    = blockIdx.x * blockDim.x + threadIdx.x;
    int stride = gridDim.x * blockDim.x;
    for (int i = idx; i < G_ * KPAD; i += stride) {
        int g = i / KPAD, k = i % KPAD;
        wih16[i] = (k < D_) ? (_Float16)wih[g * D_ + k] : (_Float16)0.0f;
    }
    for (int i = idx; i < G_ * H_; i += stride) {
        whh16[i] = (_Float16)whh[i];
    }
}

// ---------------------------------------------------------------------------
// Kernel 0b: x (n,t,d) f32 -> xp[(t*NSEQ+n)][KPAD] f16 (GEMM-M-major, padded)
// ---------------------------------------------------------------------------
__global__ void k_cvt_x(const float* __restrict__ x, _Float16* __restrict__ xp) {
    const size_t total  = (size_t)T_ * NSEQ * KPAD;
    size_t idx    = (size_t)blockIdx.x * blockDim.x + threadIdx.x;
    size_t stride = (size_t)gridDim.x * blockDim.x;
    for (size_t i = idx; i < total; i += stride) {
        int k = (int)(i % KPAD);
        size_t m = i / KPAD;              // m = t*NSEQ + n
        int t = (int)(m / NSEQ);
        int n = (int)(m % NSEQ);
        xp[i] = (k < D_) ? (_Float16)x[((size_t)n * T_ + t) * D_ + k] : (_Float16)0.0f;
    }
}

// ---------------------------------------------------------------------------
// Kernel 1: XG[t][n][g] = xp[m][:] . w_ih1[g][:] + b_ih1[g]   (f16 out)
// WMMA GEMM: M = 131072, N = 768, K = 320. Block tile 128x128, 8 waves,
// wave tile 64x32. A-tiles DMA'd by the TDM into double-buffered LDS so the
// copy of tile kc+1 overlaps the WMMAs of tile kc (s_wait_tensorcnt 1).
// ---------------------------------------------------------------------------
__global__ __launch_bounds__(256) void k_xgemm(const _Float16* __restrict__ xp,
                                               const _Float16* __restrict__ wih16,
                                               const float* __restrict__ b_ih,
                                               _Float16* __restrict__ xg) {
    __shared__ _Float16 sA[2][128][40];   // 80 B row stride (16B aligned)

    const int bx    = blockIdx.x;
    const int tileN = bx % (G_ / 128);        // 0..5
    const int tileM = bx / (G_ / 128);        // 0..1023
    const int m0    = tileM * 128;
    const int t     = m0 / NSEQ;              // whole tile shares t (128 | 2048)
    const int n0    = m0 % NSEQ;

    const int tid  = threadIdx.x;
    const int wave = tid >> 5;
    const int lane = tid & 31;
    const int l    = lane & 15;
    const int hi   = lane >> 4;
    const int wm   = wave & 1;                // rows 64*wm
    const int wn   = wave >> 1;               // cols 32*wn

    const size_t abase = (size_t)m0 * KPAD;   // element offset of tile row 0

    v8f acc[4][2];
#pragma unroll
    for (int a = 0; a < 4; ++a)
#pragma unroll
        for (int b = 0; b < 2; ++b) acc[a][b] = v8f_zero();

#if HAVE_TDM
    if (wave == 0)
        tdm_load_2d(&sA[0][0][0], (const void*)(xp + abase),
                    32u, 128u, KPAD, PADBITS_64B_PLUS_16B);
#endif

    for (int kc = 0; kc < KCH; ++kc) {
        __syncthreads();                      // prior-iter reads of sA[(kc+1)&1] done
#if HAVE_TDM
        if (wave == 0) {
            if (kc + 1 < KCH) {
                tdm_load_2d(&sA[(kc + 1) & 1][0][0],
                            (const void*)(xp + abase + (size_t)(kc + 1) * 32),
                            32u, 128u, KPAD, PADBITS_64B_PLUS_16B);
                __builtin_amdgcn_s_wait_tensorcnt(1);   // tile kc landed
            } else {
                __builtin_amdgcn_s_wait_tensorcnt(0);
            }
        }
#else
        {
            const int r  = tid >> 1;
            const int k0 = (tid & 1) * 16;
            const _Float16* src = xp + abase + (size_t)r * KPAD + kc * 32 + k0;
            *(v8h*)&sA[kc & 1][r][k0]     = *(const v8h*)src;
            *(v8h*)&sA[kc & 1][r][k0 + 8] = *(const v8h*)(src + 8);
        }
#endif
        __syncthreads();                      // tile visible to all waves

        v16h bfrag[2];
#pragma unroll
        for (int sn = 0; sn < 2; ++sn) {
            int g = tileN * 128 + wn * 32 + sn * 16 + l;
            bfrag[sn] = *(const v16h*)(wih16 + (size_t)g * KPAD + kc * 32 + hi * 16);
        }
#pragma unroll
        for (int sm = 0; sm < 4; ++sm) {
            v16h afrag = make_a_frag(&sA[kc & 1][wm * 64 + sm * 16 + l][hi * 16]);
#pragma unroll
            for (int sn = 0; sn < 2; ++sn) {
                acc[sm][sn] = __builtin_amdgcn_wmma_f32_16x16x32_f16(
                    false, afrag, false, bfrag[sn], (short)0, acc[sm][sn], false, false);
            }
        }
    }

    // epilogue: + b_ih, convert f16, scatter per C/D layout (VGPR i -> M=i+8*hi)
#pragma unroll
    for (int sm = 0; sm < 4; ++sm) {
#pragma unroll
        for (int sn = 0; sn < 2; ++sn) {
            int g = tileN * 128 + wn * 32 + sn * 16 + l;
            float bias = b_ih[g];
#pragma unroll
            for (int i = 0; i < 8; ++i) {
                int m = wm * 64 + sm * 16 + hi * 8 + i;
                size_t idx = ((size_t)t * NSEQ + (n0 + m)) * G_ + g;
                xg[idx] = (_Float16)(acc[sm][sn][i] + bias);
            }
        }
    }
}

// ---------------------------------------------------------------------------
// Kernel 2: layer-1 GRU recurrence. 64 blocks x 32 rows. h double-buffered in
// LDS (f16); per step hg = h @ w_hh.T via WMMA, fused gates. The XG slice of
// step t+1 (contiguous 48KB) is streamed into double-buffered LDS by the TDM
// while step t computes, so gate inputs come from LDS, not global.
// ---------------------------------------------------------------------------
__global__ __launch_bounds__(256) void k_gru1(const _Float16* __restrict__ xg,
                                              const _Float16* __restrict__ whh16,
                                              const float* __restrict__ b_hh,
                                              float* __restrict__ h1out) {
    __shared__ _Float16 hbuf[2][32][264];    // 33,792 B (row stride 16B aligned)
#if HAVE_TDM
    __shared__ _Float16 sX[2][32 * G_];      // 98,304 B: XG slice ping-pong
#endif

    const int n0   = blockIdx.x * 32;
    const int tid  = threadIdx.x;
    const int wave = tid >> 5;
    const int lane = tid & 31;
    const int l    = lane & 15;
    const int hi   = lane >> 4;
    const int wm   = wave & 1;
    const int grp  = wave >> 1;            // 0..3

    for (int i = tid; i < 32 * 264; i += 256)
        hbuf[0][i / 264][i % 264] = (_Float16)0.0f;

#if HAVE_TDM
    if (wave == 0)   // stage step 0's XG slice (32*768 f16, contiguous)
        tdm_load_2d(&sX[0][0], (const void*)(xg + (size_t)n0 * G_),
                    32u * G_, 1u, 32u * G_, 0u);
#endif
    __syncthreads();

    int cur = 0;
    for (int t = 0; t < T_; ++t) {
        const int nxt = cur ^ 1;
#if HAVE_TDM
        if (wave == 0) {
            if (t + 1 < T_) {
                tdm_load_2d(&sX[(t + 1) & 1][0],
                            (const void*)(xg + ((size_t)(t + 1) * NSEQ + n0) * G_),
                            32u * G_, 1u, 32u * G_, 0u);
                __builtin_amdgcn_s_wait_tensorcnt(1);   // slice t landed
            } else {
                __builtin_amdgcn_s_wait_tensorcnt(0);
            }
        }
        __syncthreads();
        const _Float16* sXc = &sX[t & 1][0];
#else
        if (t + 1 < T_) {   // no TDM: at least pull next slice toward L2
            const _Float16* pf = xg + ((size_t)(t + 1) * NSEQ + (n0 + lane)) * (size_t)G_;
            __builtin_prefetch((const void*)pf, 0, 1);
            __builtin_prefetch((const void*)(pf + 384), 0, 1);
        }
        const _Float16* sXc = xg + ((size_t)t * NSEQ + n0) * G_;   // global fallback
#endif
#pragma unroll 1
        for (int ct = 0; ct < 4; ++ct) {
            const int col = grp * 64 + ct * 16;      // within H
            v8f ar = v8f_zero(), az = v8f_zero(), an = v8f_zero();
#pragma unroll
            for (int kc = 0; kc < H_ / 32; ++kc) {
                v16h afrag = make_a_frag(&hbuf[cur][wm * 16 + l][kc * 32 + hi * 16]);
                {
                    const _Float16* bp = whh16 + (size_t)(0 * H_ + col + l) * H_ + kc * 32 + hi * 16;
                    ar = __builtin_amdgcn_wmma_f32_16x16x32_f16(false, afrag, false, *(const v16h*)bp,
                                                                (short)0, ar, false, false);
                }
                {
                    const _Float16* bp = whh16 + (size_t)(1 * H_ + col + l) * H_ + kc * 32 + hi * 16;
                    az = __builtin_amdgcn_wmma_f32_16x16x32_f16(false, afrag, false, *(const v16h*)bp,
                                                                (short)0, az, false, false);
                }
                {
                    const _Float16* bp = whh16 + (size_t)(2 * H_ + col + l) * H_ + kc * 32 + hi * 16;
                    an = __builtin_amdgcn_wmma_f32_16x16x32_f16(false, afrag, false, *(const v16h*)bp,
                                                                (short)0, an, false, false);
                }
            }
            const int c = col + l;
            const float bhr = b_hh[c], bhz = b_hh[H_ + c], bhn = b_hh[2 * H_ + c];
#pragma unroll
            for (int i = 0; i < 8; ++i) {
                const int rl = wm * 16 + hi * 8 + i;
                const size_t xb = (size_t)rl * G_;
                float xr = (float)sXc[xb + c];
                float xz = (float)sXc[xb + H_ + c];
                float xn = (float)sXc[xb + 2 * H_ + c];
                float hold = (float)hbuf[cur][rl][c];
                float r  = sig_f(xr + ar[i] + bhr);
                float z  = sig_f(xz + az[i] + bhz);
                float nn = tanh_f(xn + r * (an[i] + bhn));
                hbuf[nxt][rl][c] = (_Float16)((1.0f - z) * nn + z * hold);
            }
        }
        __syncthreads();
        cur ^= 1;
    }

    for (int i = tid; i < 32 * H_; i += 256) {
        int r = i >> 8, c = i & 255;
        h1out[(size_t)(n0 + r) * H_ + c] = (float)hbuf[cur][r][c];
    }
}

// ---------------------------------------------------------------------------
// Kernel 3: head — mid select, two zero-h GRU cells, 2-layer classifier.
// ---------------------------------------------------------------------------
__global__ __launch_bounds__(256) void k_head(const float* __restrict__ h1,
        const float* __restrict__ wih2f, const float* __restrict__ bih2f, const float* __restrict__ bhh2f,
        const float* __restrict__ wih2b, const float* __restrict__ bih2b, const float* __restrict__ bhh2b,
        const float* __restrict__ wc1, const float* __restrict__ bc1,
        const float* __restrict__ wc2, const float* __restrict__ bc2,
        float* __restrict__ out) {
    __shared__ float smid[H_];
    __shared__ float semb[2 * H_];
    __shared__ float shid[H_];

    const int bb = blockIdx.x;
    const int c  = threadIdx.x;

    smid[c] = h1[(size_t)(bb * SS_ + SS_ / 2) * H_ + c];
    __syncthreads();

    {   // forward cell (h0 == 0 -> hg = b_hh only)
        float xr = bih2f[c], xz = bih2f[H_ + c], xn = bih2f[2 * H_ + c];
        for (int k = 0; k < H_; ++k) {
            float m = smid[k];
            xr += m * wih2f[(size_t)c * H_ + k];
            xz += m * wih2f[(size_t)(H_ + c) * H_ + k];
            xn += m * wih2f[(size_t)(2 * H_ + c) * H_ + k];
        }
        float r  = sig_f(xr + bhh2f[c]);
        float z  = sig_f(xz + bhh2f[H_ + c]);
        float nn = tanh_f(xn + r * bhh2f[2 * H_ + c]);
        semb[c] = (1.0f - z) * nn;
    }
    {   // backward cell
        float xr = bih2b[c], xz = bih2b[H_ + c], xn = bih2b[2 * H_ + c];
        for (int k = 0; k < H_; ++k) {
            float m = smid[k];
            xr += m * wih2b[(size_t)c * H_ + k];
            xz += m * wih2b[(size_t)(H_ + c) * H_ + k];
            xn += m * wih2b[(size_t)(2 * H_ + c) * H_ + k];
        }
        float r  = sig_f(xr + bhh2b[c]);
        float z  = sig_f(xz + bhh2b[H_ + c]);
        float nn = tanh_f(xn + r * bhh2b[2 * H_ + c]);
        semb[H_ + c] = (1.0f - z) * nn;
    }
    __syncthreads();

    float pre = bc1[c];
    for (int k = 0; k < 2 * H_; ++k) pre += semb[k] * wc1[(size_t)c * (2 * H_) + k];
    shid[c] = (pre >= 0.0f) ? pre : 0.1f * pre;
    __syncthreads();

    if (c < 2) {
        float acc = bc2[c];
        for (int k = 0; k < H_; ++k) acc += shid[k] * wc2[(size_t)c * H_ + k];
        out[bb * 2 + c] = acc;
    }
}

// ---------------------------------------------------------------------------
extern "C" void kernel_launch(void* const* d_in, const int* in_sizes, int n_in,
                              void* d_out, int out_size, void* d_ws, size_t ws_size,
                              hipStream_t stream) {
    const float* x      = (const float*)d_in[0];
    const float* w_ih1  = (const float*)d_in[1];
    const float* w_hh1  = (const float*)d_in[2];
    const float* b_ih1  = (const float*)d_in[3];
    const float* b_hh1  = (const float*)d_in[4];
    const float* w_ih2f = (const float*)d_in[5];
    const float* b_ih2f = (const float*)d_in[7];
    const float* b_hh2f = (const float*)d_in[8];
    const float* w_ih2b = (const float*)d_in[9];
    const float* b_ih2b = (const float*)d_in[11];
    const float* b_hh2b = (const float*)d_in[12];
    const float* wc1    = (const float*)d_in[13];
    const float* bc1    = (const float*)d_in[14];
    const float* wc2    = (const float*)d_in[15];
    const float* bc2    = (const float*)d_in[16];

    char* ws = (char*)d_ws;
    const size_t XG_BYTES  = (size_t)T_ * NSEQ * G_ * sizeof(_Float16);    // 201,326,592
    const size_t XP_BYTES  = (size_t)T_ * NSEQ * KPAD * sizeof(_Float16); //  83,886,080
    const size_t WIH_BYTES = (size_t)G_ * KPAD * sizeof(_Float16);        //     491,520
    const size_t WHH_BYTES = (size_t)G_ * H_ * sizeof(_Float16);          //     393,216

    _Float16* xg    = (_Float16*)ws;
    _Float16* xp    = (_Float16*)(ws + XG_BYTES);
    _Float16* wih16 = (_Float16*)(ws + XG_BYTES + XP_BYTES);
    _Float16* whh16 = (_Float16*)(ws + XG_BYTES + XP_BYTES + WIH_BYTES);
    float*    h1    = (float*)   (ws + XG_BYTES + XP_BYTES + WIH_BYTES + WHH_BYTES);

    k_cvt_weights<<<512, 256, 0, stream>>>(w_ih1, w_hh1, wih16, whh16);
    k_cvt_x<<<4096, 256, 0, stream>>>(x, xp);
    k_xgemm<<<(NSEQ * T_ / 128) * (G_ / 128), 256, 0, stream>>>(xp, wih16, b_ih1, xg);
    k_gru1<<<NSEQ / 32, 256, 0, stream>>>(xg, whh16, b_hh1, h1);
    k_head<<<B_, 256, 0, stream>>>(h1, w_ih2f, b_ih2f, b_hh2f,
                                   w_ih2b, b_ih2b, b_hh2b,
                                   wc1, bc1, wc2, bc2, (float*)d_out);
}
